// NeuralNetwork7_82325933130163
// MI455X (gfx1250) — matrix operand order repaired and checked by reading, not compiled, not run on hardware
//
#include <hip/hip_runtime.h>

typedef __attribute__((ext_vector_type(16))) _Float16 v16h;
typedef __attribute__((ext_vector_type(8)))  float    v8f;

#define NTOK   262144
#define DDIM   128
#define NVQS   6
#define KCODES 512
#define ROWS_PER_WAVE 32
#define WAVES_PB 2
#define NWTILES (NTOK / ROWS_PER_WAVE)     // 8192 waves, 32 rows each
#define RPITCH 132                          // padded row pitch (floats), 16B-aligned

// workspace layout (bytes)
#define WS_WF16_OFF   0                         // 128*128 f16   = 32768
#define WS_CB16_OFF   32768                     // 6*512*128 f16 = 786432
#define WS_NORM_OFF   (32768 + 786432)          // 6*512 f32     = 12288
#define WS_CPART_OFF  (32768 + 786432 + 12288)  // 8192 f32      = 32768
// NOTE: the double-buffered prefetch overreads <= ~4.2KB past the stage-5
// codebook / norms; those bytes land inside the norms/cparts regions above,
// i.e. still inside our own workspace. Values are never used.

// ---------------- prep: f32->f16 codebooks + ||E||^2, f32->f16 weight ----------
__global__ __launch_bounds__(128) void prep_kernel(
    const float* __restrict__ cb, const float* __restrict__ W,
    _Float16* __restrict__ Wf16, _Float16* __restrict__ cbf16,
    float* __restrict__ norms)
{
  int b = blockIdx.x, t = threadIdx.x;
  if (b < NVQS*KCODES) {
    __shared__ float red[128];
    float e = cb[(size_t)b*DDIM + t];
    cbf16[(size_t)b*DDIM + t] = (_Float16)e;
    red[t] = e*e;
    __syncthreads();
    #pragma unroll
    for (int s = 64; s > 0; s >>= 1) { if (t < s) red[t] += red[t+s]; __syncthreads(); }
    if (t == 0) norms[b] = red[0];
  } else {
    int r = b - NVQS*KCODES;               // 0..127 : weight rows
    Wf16[r*DDIM + t] = (_Float16)W[r*DDIM + t];
  }
}

// ---------------- deterministic commit reduction ------------------------------
__global__ __launch_bounds__(256) void commit_reduce_kernel(
    const float* __restrict__ parts, float* __restrict__ out)
{
  __shared__ float red[256];
  int t = threadIdx.x;
  float s = 0.f;
  for (int i = t; i < NWTILES; i += 256) s += parts[i];
  red[t] = s;
  __syncthreads();
  #pragma unroll
  for (int k = 128; k > 0; k >>= 1) { if (t < k) red[t] += red[t+k]; __syncthreads(); }
  if (t == 0) *out = red[0] * (1.0f / ((float)NTOK * (float)DDIM));
}

// Build one 16x32 f16 A-fragment chunk from an LDS residual row.
// Span order: elems 0-15 = K[16h,16h+16) (+32c).
__device__ __forceinline__ v16h buildA(const float* Rrow, int c, int half)
{
  const float* p = Rrow + 32*c + 16*half;
  float4 a0 = *(const float4*)(p);
  float4 a1 = *(const float4*)(p + 4);
  float4 a2 = *(const float4*)(p + 8);
  float4 a3 = *(const float4*)(p + 12);
  v16h A;
  A[0]=(_Float16)a0.x;  A[1]=(_Float16)a0.y;  A[2]=(_Float16)a0.z;  A[3]=(_Float16)a0.w;
  A[4]=(_Float16)a1.x;  A[5]=(_Float16)a1.y;  A[6]=(_Float16)a1.z;  A[7]=(_Float16)a1.w;
  A[8]=(_Float16)a2.x;  A[9]=(_Float16)a2.y;  A[10]=(_Float16)a2.z; A[11]=(_Float16)a2.w;
  A[12]=(_Float16)a3.x; A[13]=(_Float16)a3.y; A[14]=(_Float16)a3.z; A[15]=(_Float16)a3.w;
  return A;
}

// load one 128x16 B tile (4 k-chunks, 32B per lane per chunk)
__device__ __forceinline__ void loadB(const _Float16* __restrict__ bp, v16h* B)
{
  #pragma unroll
  for (int c = 0; c < 4; ++c) B[c] = *(const v16h*)(bp + 32*c);
}

// 8 WMMAs (two independent chains) + argmin fold for one 16-code column tile
__device__ __forceinline__ void compute_tile(
    const v16h* A0, const v16h* A1, const v16h* B,
    float nrm, int code,
    float* bd0, int* bi0, float* bd1, int* bi1)
{
  v8f acc0 = {}; v8f acc1 = {};
  #pragma unroll
  for (int c = 0; c < 4; ++c) {
    acc0 = __builtin_amdgcn_wmma_f32_16x16x32_f16(false, A0[c], false, B[c],
                                                  (short)0, acc0, false, false);
    acc1 = __builtin_amdgcn_wmma_f32_16x16x32_f16(false, A1[c], false, B[c],
                                                  (short)0, acc1, false, false);
  }
  #pragma unroll
  for (int r = 0; r < 8; ++r) {
    float d0 = nrm - 2.0f * acc0[r];         // ||E||^2 - 2 x.E  (||x||^2 const)
    float d1 = nrm - 2.0f * acc1[r];
    if (d0 < bd0[r]) { bd0[r] = d0; bi0[r] = code; }
    if (d1 < bd1[r]) { bd1[r] = d1; bi1[r] = code; }
  }
}

// ---------------- fused 6-stage RVQ + linear, 32 rows per wave ----------------
__global__ __launch_bounds__(WAVES_PB*32) void rvq_fused_kernel(
    const float* __restrict__ x, const float* __restrict__ cb,
    const float* __restrict__ lin_b,
    const _Float16* __restrict__ Wf16, const _Float16* __restrict__ cbf16,
    const float* __restrict__ norms,
    float* __restrict__ outY, int* __restrict__ outIdx,
    float* __restrict__ commitParts)
{
  __shared__ float resLDS[WAVES_PB * ROWS_PER_WAVE * RPITCH];
  __shared__ int   idxLDS[WAVES_PB * ROWS_PER_WAVE];

  const int lane  = threadIdx.x & 31;
  const int wave  = threadIdx.x >> 5;
  const int wtile = blockIdx.x * WAVES_PB + wave;
  const int rowBase = wtile * ROWS_PER_WAVE;
  const int m    = lane & 15;    // A row (within tile) / D column / B code-column
  const int half = lane >> 4;

  float* R    = resLDS + wave * ROWS_PER_WAVE * RPITCH;
  int*   widx = idxLDS + wave * ROWS_PER_WAVE;

  // stage residual = x into LDS, one row per lane
  {
    const float* xr0 = x + (size_t)(rowBase + lane) * DDIM;
    float* dst = R + lane * RPITCH;
    #pragma unroll
    for (int d = 0; d < DDIM; d += 4)
      *(float4*)(dst + d) = *(const float4*)(xr0 + d);
  }
  asm volatile("s_wait_dscnt 0" ::: "memory");

  float commitSum = 0.f;

  #pragma unroll 1
  for (int s = 0; s < NVQS; ++s) {
    // A fragments for both 16-row tiles (lane m -> rows m and m+16)
    const float* R0 = R + m * RPITCH;
    const float* R1 = R + (m + 16) * RPITCH;
    v16h A0[4], A1[4];
    #pragma unroll
    for (int c = 0; c < 4; ++c) { A0[c] = buildA(R0, c, half); A1[c] = buildA(R1, c, half); }

    const _Float16* cbs = cbf16 + (size_t)s * KCODES * DDIM;
    const float*    ns  = norms + s * KCODES;
    const _Float16* colBase = cbs + (size_t)m * DDIM + (size_t)(16*half);

    float bd0[8], bd1[8]; int bi0[8], bi1[8];
    #pragma unroll
    for (int r = 0; r < 8; ++r) { bd0[r] = 3.4e38f; bi0[r] = 0; bd1[r] = 3.4e38f; bi1[r] = 0; }

    // 512 codes = 32 column tiles; double-buffered B stream, each B fragment
    // feeds both row tiles (two independent WMMA chains)
    v16h Ba[4], Bb[4];
    loadB(colBase, Ba);
    float nA = ns[m];
    #pragma unroll 1
    for (int ct = 0; ct < 32; ct += 2) {
      loadB(colBase + (size_t)(ct + 1) * 16 * DDIM, Bb);     // prefetch tile ct+1
      float nB = ns[(ct + 1) * 16 + m];
      compute_tile(A0, A1, Ba, nA, ct * 16 + m, bd0, bi0, bd1, bi1);

      loadB(colBase + (size_t)(ct + 2) * 16 * DDIM, Ba);     // prefetch tile ct+2
      nA = ns[(ct + 2) * 16 + m];                            // (ws-internal overread at ct=30)
      compute_tile(A0, A1, Bb, nB, (ct + 1) * 16 + m, bd0, bi0, bd1, bi1);
    }

    // min across the 16 column lanes of each half-group (ties -> lower index)
    #pragma unroll
    for (int r = 0; r < 8; ++r) {
      #pragma unroll
      for (int off = 1; off < 16; off <<= 1) {
        float od0 = __shfl_xor(bd0[r], off, 32);
        int   oi0 = __shfl_xor(bi0[r], off, 32);
        if (od0 < bd0[r] || (od0 == bd0[r] && oi0 < bi0[r])) { bd0[r] = od0; bi0[r] = oi0; }
        float od1 = __shfl_xor(bd1[r], off, 32);
        int   oi1 = __shfl_xor(bi1[r], off, 32);
        if (od1 < bd1[r] || (od1 == bd1[r] && oi1 < bi1[r])) { bd1[r] = od1; bi1[r] = oi1; }
      }
    }

    // publish per-row winning index (rows 8h+r tile0, 16+8h+r tile1)
    if (m == 0) {
      #pragma unroll
      for (int r = 0; r < 8; ++r) {
        widx[half*8 + r]      = bi0[r];
        widx[16 + half*8 + r] = bi1[r];
      }
    }
    asm volatile("s_wait_dscnt 0" ::: "memory");

    // gather q = E[idx] (f32 master codebook); one full row per lane
    int myIdx = widx[lane];
    if (s == 0) outIdx[rowBase + lane] = myIdx;

    const float* q = cb + ((size_t)(s * KCODES) + myIdx) * DDIM;
    float* Rrow = R + lane * RPITCH;
    #pragma unroll
    for (int d = 0; d < DDIM; d += 4) {
      float4 qv = *(const float4*)(q + d);
      float4 rv = *(const float4*)(Rrow + d);
      rv.x -= qv.x; rv.y -= qv.y; rv.z -= qv.z; rv.w -= qv.w;
      *(float4*)(Rrow + d) = rv;
      if (s == 0)
        commitSum += rv.x*rv.x + rv.y*rv.y + rv.z*rv.z + rv.w*rv.w;
    }
    asm volatile("s_wait_dscnt 0" ::: "memory");
  }

  // commit partial (stage 0): fixed-order wave reduce -> one slot per wave tile
  #pragma unroll
  for (int off = 1; off < 32; off <<= 1) commitSum += __shfl_xor(commitSum, off, 32);
  if (lane == 0) commitParts[wtile] = commitSum;

  // y = x_input - residual_final (sum of quantized); write back into LDS
  {
    const float* xr0 = x + (size_t)(rowBase + lane) * DDIM;
    float* Rrow = R + lane * RPITCH;
    #pragma unroll
    for (int d = 0; d < DDIM; d += 4) {
      float4 xv = *(const float4*)(xr0 + d);
      float4 rv = *(const float4*)(Rrow + d);
      rv.x = xv.x - rv.x; rv.y = xv.y - rv.y; rv.z = xv.z - rv.z; rv.w = xv.w - rv.w;
      *(float4*)(Rrow + d) = rv;
    }
  }
  asm volatile("s_wait_dscnt 0" ::: "memory");

  // final linear: out = y * W^T + b, two 16-row tiles per wave
  {
    const float* R0 = R + m * RPITCH;
    const float* R1 = R + (m + 16) * RPITCH;
    v16h Ay0[4], Ay1[4];
    #pragma unroll
    for (int c = 0; c < 4; ++c) { Ay0[c] = buildA(R0, c, half); Ay1[c] = buildA(R1, c, half); }

    for (int ot = 0; ot < 8; ++ot) {
      int o0 = ot * 16;
      const _Float16* wrow = Wf16 + (size_t)(o0 + m) * DDIM + 16*half;
      v8f acc0 = {}; v8f acc1 = {};
      #pragma unroll
      for (int c = 0; c < 4; ++c) {
        v16h B = *(const v16h*)(wrow + 32*c);
        acc0 = __builtin_amdgcn_wmma_f32_16x16x32_f16(false, Ay0[c], false, B,
                                                      (short)0, acc0, false, false);
        acc1 = __builtin_amdgcn_wmma_f32_16x16x32_f16(false, Ay1[c], false, B,
                                                      (short)0, acc1, false, false);
      }
      float bias = lin_b[o0 + m];
      #pragma unroll
      for (int r = 0; r < 8; ++r) {
        int row0 = half*8 + r;
        outY[(size_t)(rowBase + row0) * DDIM + o0 + m]      = acc0[r] + bias;
        outY[(size_t)(rowBase + 16 + row0) * DDIM + o0 + m] = acc1[r] + bias;
      }
    }
  }
}

extern "C" void kernel_launch(void* const* d_in, const int* in_sizes, int n_in,
                              void* d_out, int out_size, void* d_ws, size_t ws_size,
                              hipStream_t stream)
{
  (void)in_sizes; (void)n_in; (void)out_size; (void)ws_size;
  const float* x    = (const float*)d_in[0];   // [N,128] f32
  const float* cb   = (const float*)d_in[1];   // [6,512,128] f32
  const float* W    = (const float*)d_in[2];   // [128,128] f32 (o,d)
  const float* bvec = (const float*)d_in[3];   // [128] f32

  float* outY      = (float*)d_out;                          // N*128 f32
  int*   outIdx    = (int*)(outY + (size_t)NTOK * DDIM);     // N int32
  float* outCommit = outY + (size_t)NTOK * DDIM + NTOK;      // 1 f32

  char* ws = (char*)d_ws;
  _Float16* Wf16   = (_Float16*)(ws + WS_WF16_OFF);
  _Float16* cbf16  = (_Float16*)(ws + WS_CB16_OFF);
  float*    norms  = (float*)(ws + WS_NORM_OFF);
  float*    cparts = (float*)(ws + WS_CPART_OFF);

  prep_kernel<<<NVQS*KCODES + 128, 128, 0, stream>>>(cb, W, Wf16, cbf16, norms);
  rvq_fused_kernel<<<NWTILES/WAVES_PB, WAVES_PB*32, 0, stream>>>(
      x, cb, bvec, Wf16, cbf16, norms, outY, outIdx, cparts);
  commit_reduce_kernel<<<1, 256, 0, stream>>>(cparts, outCommit);
}
